// PLanetoidGCN1_27977416966231
// MI455X (gfx1250) — compile-verified, matrix operand order checked by reading, not run on hardware
//
#include <hip/hip_runtime.h>
#include <hip/hip_bf16.h>

typedef __attribute__((ext_vector_type(16))) _Float16 v16h;
typedef __attribute__((ext_vector_type(8)))  float    v8f;

#define N_NODES 100000
#define N_EDGES 1600000
#define D_IN    512
#define D_OUT   256

// ---------------------------------------------------------------- degree ----
__global__ void k_init_deg(float* __restrict__ deg) {
    int i = blockIdx.x * blockDim.x + threadIdx.x;
    if (i < N_NODES) deg[i] = 1.0f;  // self loop
}

__global__ void k_deg(const long long* __restrict__ ei, float* __restrict__ deg) {
    int e = blockIdx.x * blockDim.x + threadIdx.x;
    if (e < N_EDGES) {
        int d = (int)ei[N_EDGES + e];   // dst row of edge_index
        atomicAdd(&deg[d], 1.0f);
    }
}

__global__ void k_dinv(const float* __restrict__ deg, float* __restrict__ dinv) {
    int i = blockIdx.x * blockDim.x + threadIdx.x;
    if (i < N_NODES) dinv[i] = rsqrtf(deg[i]);   // deg >= 1 always
}

// ------------------------------------------------- W -> f16, transposed -----
// Wt[n][k] = (f16) W[k][n]  so B fragments are contiguous along K.
__global__ void k_wt(const float* __restrict__ W, _Float16* __restrict__ Wt) {
    int t = blockIdx.x * blockDim.x + threadIdx.x;     // 256*512 elements
    if (t < D_IN * D_OUT) {
        int n = t >> 9;          // /512
        int k = t & (D_IN - 1);  // %512
        Wt[t] = (_Float16)W[k * D_OUT + n];
    }
}

// ----------------------------------------------------------------- GEMM -----
// h[N,256] (f16) = x[N,512] @ W[512,256], f16 inputs / f32 accumulate.
// One wave32 -> one 16x128 strip: A fragment loaded once, reused across
// 8 column tiles (8 independent WMMA accumulator chains = 64 acc VGPRs).
__global__ void __launch_bounds__(128)
k_gemm(const float* __restrict__ x, const _Float16* __restrict__ Wt,
       _Float16* __restrict__ h) {
    const int lane   = threadIdx.x & 31;
    const int wave   = threadIdx.x >> 5;
    const int tile   = blockIdx.x * 4 + wave;   // 3125 blocks * 4 waves = 12500 strips
    const int rt     = tile >> 1;               // 6250 row tiles (100000/16 exact)
    const int cg     = tile & 1;                // column group: cols [cg*128, cg*128+128)
    const int half   = lane >> 4;               // K sub-block select (ISA A/B layout)
    const int lane16 = lane & 15;

    const int m = rt * 16 + lane16;             // A row for this lane
    const float*    xrow  = x  + (size_t)m * D_IN;
    const _Float16* wbase = Wt + (size_t)(cg * 128 + lane16) * D_IN;

    v8f acc[8];
    #pragma unroll
    for (int c = 0; c < 8; ++c) acc[c] = (v8f){};

    #pragma unroll 1
    for (int ks = 0; ks < D_IN; ks += 32) {
        const int kb = ks + half * 8;
        v16h a;
        // A 16x32 f16 layout: elems 0..7 -> K=kb..kb+7, elems 8..15 -> K=kb+16..kb+23
        #pragma unroll
        for (int j = 0; j < 8; ++j) {
            a[j]     = (_Float16)xrow[kb + j];
            a[j + 8] = (_Float16)xrow[kb + 16 + j];
        }
        // 8 column tiles share the same A fragment
        #pragma unroll
        for (int c = 0; c < 8; ++c) {
            const _Float16* wrow = wbase + (size_t)c * 16 * D_IN;
            v16h b;
            #pragma unroll
            for (int j = 0; j < 8; ++j) {
                b[j]     = wrow[kb + j];
                b[j + 8] = wrow[kb + 16 + j];
            }
            acc[c] = __builtin_amdgcn_wmma_f32_16x16x32_f16(
                /*neg_a=*/false, a, /*neg_b=*/false, b,
                /*c_mod=*/(short)0, acc[c], /*reuse_a=*/false, /*reuse_b=*/false);
        }
    }

    // D layout: VGPR v -> M = v (+8 for lanes 16..31), N = lane16 within tile
    const int mbase = rt * 16 + half * 8;
    #pragma unroll
    for (int c = 0; c < 8; ++c) {
        const int n = cg * 128 + c * 16 + lane16;
        _Float16* hp = h + (size_t)mbase * D_OUT + n;
        #pragma unroll
        for (int v = 0; v < 8; ++v) hp[(size_t)v * D_OUT] = (_Float16)acc[c][v];
    }
}

// ----------------------------------------- out = b + (1/deg_i) * h_i --------
// self-loop contribution: norm_ii = dinv[i]*dinv[i] = 1/deg[i]
__global__ void k_out_init(const _Float16* __restrict__ h, const float* __restrict__ dinv,
                           const float* __restrict__ bias, float* __restrict__ out) {
    const int i = blockIdx.x;        // node
    const int j = threadIdx.x;       // feature (256)
    const size_t t = (size_t)i * D_OUT + j;
    const float di = dinv[i];
    out[t] = bias[j] + (float)h[t] * di * di;
}

// -------------------------------------------------- edge scatter-add --------
// One block per edge, 256 lanes = one feature row; f16 gather halves the
// dominant read stream (0.8 GB); f32 atomics land in the 192MB L2 where the
// 102.4MB `out` buffer is resident.
__global__ void k_scatter(const long long* __restrict__ ei, const _Float16* __restrict__ h,
                          const float* __restrict__ dinv, float* __restrict__ out) {
    const int e = blockIdx.x;
    const int s = (int)ei[e];
    const int d = (int)ei[N_EDGES + e];
    const float w = dinv[s] * dinv[d];
    const int j = threadIdx.x;
    atomicAdd(&out[(size_t)d * D_OUT + j], (float)h[(size_t)s * D_OUT + j] * w);
}

// ----------------------------------------------------------- PReLU ----------
__global__ void k_prelu(float* __restrict__ out, const float* __restrict__ pa) {
    const size_t t = (size_t)blockIdx.x * blockDim.x + threadIdx.x;
    if (t < (size_t)N_NODES * D_OUT) {
        const float a = pa[0];
        const float v = out[t];
        out[t] = fmaxf(v, 0.0f) + a * fminf(v, 0.0f);
    }
}

// ---------------------------------------------------------------------------
extern "C" void kernel_launch(void* const* d_in, const int* in_sizes, int n_in,
                              void* d_out, int out_size, void* d_ws, size_t ws_size,
                              hipStream_t stream) {
    const float*     x  = (const float*)d_in[0];
    const long long* ei = (const long long*)d_in[1];   // edge_index int64 [2,E]
    const float*     W  = (const float*)d_in[2];
    const float*     b  = (const float*)d_in[3];
    const float*     pa = (const float*)d_in[4];
    float* out = (float*)d_out;

    // workspace layout (aligned offsets), total ~52.4 MB
    char* ws = (char*)d_ws;
    float*    deg  = (float*)   (ws + 0);          //   400,000 B
    float*    dinv = (float*)   (ws + 401408);     //   400,000 B
    _Float16* Wt   = (_Float16*)(ws + 802816);     //   262,144 B
    _Float16* h    = (_Float16*)(ws + 1114112);    //  51,200,000 B (f16)

    k_init_deg<<<(N_NODES + 255) / 256, 256, 0, stream>>>(deg);
    k_deg     <<<(N_EDGES + 255) / 256, 256, 0, stream>>>(ei, deg);
    k_dinv    <<<(N_NODES + 255) / 256, 256, 0, stream>>>(deg, dinv);
    k_wt      <<<(D_IN * D_OUT) / 256, 256, 0, stream>>>(W, Wt);
    k_gemm    <<<3125, 128, 0, stream>>>(x, Wt, h);
    k_out_init<<<N_NODES, D_OUT, 0, stream>>>(h, dinv, b, out);
    k_scatter <<<N_EDGES, D_OUT, 0, stream>>>(ei, h, dinv, out);
    k_prelu   <<<((size_t)N_NODES * D_OUT + 255) / 256, 256, 0, stream>>>(out, pa);
}
